// GraphEncoder_13417477833490
// MI455X (gfx1250) — matrix operand order repaired and checked by reading, not compiled, not run on hardware
//
#include <hip/hip_runtime.h>
#include <hip/hip_bf16.h>

// ---------------------------------------------------------------------------
// GCN encoder for MI455X / gfx1250 (wave32, WMMA).
//
// Strategy:
//  * One-time vectorized conversions: x -> bf16 row-major (xbf),
//    W1 -> bf16 transposed [96][512] (W1t), [W_mu|W_logstd] -> bf16
//    transposed [64][96] (WcT), hidden -> bf16 row-major (Hbf).
//  * GEMMs run LDS-free: per the CDNA5 16-bit WMMA VGPR layouts, each lane's
//    A fragment is two contiguous 16B row chunks and each lane's B fragment
//    is two contiguous 16B column chunks (from the transposed weights), so
//    fragments come straight from global_load_b128 and feed
//    v_wmma_f32_16x16x32_bf16 with fp32 accumulation.
//  * Edge aggregation: one wave per edge, lanes sweep features; fp32 atomics
//    resolve in the 192MB L2 (H1=19MB, H2=13MB working sets are L2-resident).
// ---------------------------------------------------------------------------

typedef __bf16 v8bf  __attribute__((ext_vector_type(8)));
typedef __bf16 v16bf __attribute__((ext_vector_type(16)));
typedef float  v8f   __attribute__((ext_vector_type(8)));

#define N_NODES 50000
#define N_IN    512
#define N_HID   96
#define N_LAT   32
#define N_OUT2  64   // [W_mu | W_logstd] fused

// ---------------- one-time bf16 conversions --------------------------------
// x: float4 in -> 4 bf16 (2 dwords) out, fully coalesced.
__global__ void k_cvt_x(const float4* __restrict__ X4,
                        unsigned int* __restrict__ Xb, long long n4) {
    long long idx = (long long)blockIdx.x * blockDim.x + threadIdx.x;
    if (idx >= n4) return;
    float4 v = X4[idx];
    union { __bf16 b[4]; unsigned int u[2]; } p;
    p.b[0] = (__bf16)v.x; p.b[1] = (__bf16)v.y;
    p.b[2] = (__bf16)v.z; p.b[3] = (__bf16)v.w;
    Xb[idx * 2]     = p.u[0];
    Xb[idx * 2 + 1] = p.u[1];
}

// W1[512][96] -> W1t[96][512] bf16 (N-major, K contiguous)
__global__ void k_cvt_w1t(const float* __restrict__ W1, __bf16* __restrict__ Wt) {
    int idx = blockIdx.x * blockDim.x + threadIdx.x;
    if (idx >= N_HID * N_IN) return;
    int n = idx >> 9, k = idx & (N_IN - 1);
    Wt[(size_t)n * N_IN + k] = (__bf16)W1[(size_t)k * N_HID + n];
}

// [W_mu|W_logstd] -> WcT[64][96] bf16 (N-major, K contiguous)
__global__ void k_cvt_wcat(const float* __restrict__ Wmu, const float* __restrict__ Wls,
                           __bf16* __restrict__ Wc) {
    int n = blockIdx.x;      // 64
    int k = threadIdx.x;     // 96
    float w = (n < N_LAT) ? Wmu[(size_t)k * N_LAT + n]
                          : Wls[(size_t)k * N_LAT + (n - N_LAT)];
    Wc[(size_t)n * N_HID + k] = (__bf16)w;
}

// ---------------- degree / normalization -----------------------------------
__global__ void k_init_deg(float* __restrict__ deg) {
    int i = blockIdx.x * blockDim.x + threadIdx.x;
    if (i < N_NODES) deg[i] = 1.0f;   // self-loop
}

__global__ void k_deg_scatter(const long long* __restrict__ dst, int E,
                              float* __restrict__ deg) {
    int e = blockIdx.x * blockDim.x + threadIdx.x;
    if (e < E) atomicAdd(&deg[(int)dst[e]], 1.0f);
}

__global__ void k_rsqrt_inplace(float* __restrict__ deg) {
    int i = blockIdx.x * blockDim.x + threadIdx.x;
    if (i < N_NODES) deg[i] = rsqrtf(deg[i]);
}

// ---------------- GEMM1: H1[50000x96] = xbf @ W1t^T -------------------------
// grid 3125 row tiles of 16, block 192 = 6 waves (one 16-col tile each).
// LDS-free: fragments loaded directly with global_load_b128.
__global__ __launch_bounds__(192)
void k_gemm1(const __bf16* __restrict__ Xb, const __bf16* __restrict__ Wt,
             float* __restrict__ H1) {
    const int tid  = threadIdx.x;
    const int wave = tid >> 5;
    const int lane = tid & 31;
    const int row0 = blockIdx.x * 16;
    const int n0   = wave * 16;
    const int ln   = lane & 15;   // A row M / B col N / C col N
    const int half = lane >> 4;

    const __bf16* arow = Xb + (size_t)(row0 + ln) * N_IN;
    const __bf16* brow = Wt + (size_t)(n0 + ln) * N_IN;

    v8f c = {0.f, 0.f, 0.f, 0.f, 0.f, 0.f, 0.f, 0.f};
#pragma unroll 4
    for (int k0 = 0; k0 < N_IN; k0 += 32) {
        // A 16x32: lane holds row ln; K chunks [k0+8h, +8) and [k0+16+8h, +8)
        union { v8bf h[2]; v16bf v; } a;
        a.h[0] = *(const v8bf*)(arow + k0 + half * 8);
        a.h[1] = *(const v8bf*)(arow + k0 + 16 + half * 8);
        // B 32x16: lane holds col ln; K chunks [k0+8h, +8) and [k0+16+8h, +8)
        union { v8bf h[2]; v16bf v; } bb;
        bb.h[0] = *(const v8bf*)(brow + k0 + half * 8);
        bb.h[1] = *(const v8bf*)(brow + k0 + 16 + half * 8);
        v16bf b = bb.v;
        c = __builtin_amdgcn_wmma_f32_16x16x32_bf16(
                false, a.v, false, b, (short)0, c, false, false);
    }
    // C/D: VGPR v -> M = v + 8*half, N = ln
#pragma unroll
    for (int v = 0; v < 8; ++v)
        H1[(size_t)(row0 + v + half * 8) * N_HID + n0 + ln] = c[v];
}

// ---------------- layer-1 self-loop + bias ----------------------------------
__global__ void k_self1(const float* __restrict__ H1, const float* __restrict__ dis,
                        const float* __restrict__ b1, float* __restrict__ A1) {
    int i = blockIdx.x;        // node
    int f = threadIdx.x;       // 96 features
    float d = dis[i];
    size_t o = (size_t)i * N_HID + f;
    A1[o] = H1[o] * d * d + b1[f];
}

// ---------------- layer-1 edge scatter: one wave per edge -------------------
__global__ __launch_bounds__(256)
void k_edge1(const long long* __restrict__ src, const long long* __restrict__ dst,
             int E, const float* __restrict__ H1, const float* __restrict__ dis,
             float* __restrict__ A1) {
    int wave = threadIdx.x >> 5;
    int lane = threadIdx.x & 31;
    int e = blockIdx.x * 8 + wave;
    if (e >= E) return;
    int s = (int)src[e], d = (int)dst[e];
    float norm = dis[s] * dis[d];
    const float* hs = H1 + (size_t)s * N_HID;
    float*       ad = A1 + (size_t)d * N_HID;
#pragma unroll
    for (int it = 0; it < 3; ++it) {
        int f = lane + it * 32;
        atomicAdd(&ad[f], hs[f] * norm);
    }
}

// ---------------- ReLU + convert hidden to bf16 (packed dwords) -------------
__global__ void k_relu_cvt(const float* __restrict__ A1, unsigned int* __restrict__ Hb) {
    int i = blockIdx.x;        // node
    int t = threadIdx.x;       // 48 pairs
    size_t o = (size_t)i * N_HID + 2 * t;
    union { __bf16 b[2]; unsigned int u; } p;
    p.b[0] = (__bf16)fmaxf(A1[o], 0.0f);
    p.b[1] = (__bf16)fmaxf(A1[o + 1], 0.0f);
    Hb[(size_t)i * (N_HID / 2) + t] = p.u;
}

// ---------------- GEMM2: H2[50000x64] = Hbf @ WcT^T -------------------------
// grid 3125, block 128 = 4 waves; K = 96 (3 steps), fully unrolled.
__global__ __launch_bounds__(128)
void k_gemm2(const __bf16* __restrict__ Hb, const __bf16* __restrict__ Wc,
             float* __restrict__ H2) {
    const int tid  = threadIdx.x;
    const int wave = tid >> 5;
    const int lane = tid & 31;
    const int row0 = blockIdx.x * 16;
    const int n0   = wave * 16;
    const int ln   = lane & 15;
    const int half = lane >> 4;

    const __bf16* arow = Hb + (size_t)(row0 + ln) * N_HID;
    const __bf16* brow = Wc + (size_t)(n0 + ln) * N_HID;

    v8f c = {0.f, 0.f, 0.f, 0.f, 0.f, 0.f, 0.f, 0.f};
#pragma unroll
    for (int k0 = 0; k0 < N_HID; k0 += 32) {
        union { v8bf h[2]; v16bf v; } a;
        a.h[0] = *(const v8bf*)(arow + k0 + half * 8);
        a.h[1] = *(const v8bf*)(arow + k0 + 16 + half * 8);
        union { v8bf h[2]; v16bf v; } bb;
        bb.h[0] = *(const v8bf*)(brow + k0 + half * 8);
        bb.h[1] = *(const v8bf*)(brow + k0 + 16 + half * 8);
        v16bf b = bb.v;
        c = __builtin_amdgcn_wmma_f32_16x16x32_bf16(
                false, a.v, false, b, (short)0, c, false, false);
    }
#pragma unroll
    for (int v = 0; v < 8; ++v)
        H2[(size_t)(row0 + v + half * 8) * N_OUT2 + n0 + ln] = c[v];
}

// ---------------- layer-2 self-loop + bias into d_out -----------------------
// d_out = mu[50000x32] then logstd[50000x32]
__global__ void k_self2(const float* __restrict__ H2, const float* __restrict__ dis,
                        const float* __restrict__ bmu, const float* __restrict__ bls,
                        float* __restrict__ out) {
    int i = blockIdx.x;        // node
    int f = threadIdx.x;       // 64
    float d = dis[i];
    float v = H2[(size_t)i * N_OUT2 + f] * d * d;
    if (f < N_LAT) out[(size_t)i * N_LAT + f] = v + bmu[f];
    else out[(size_t)N_NODES * N_LAT + (size_t)i * N_LAT + (f - N_LAT)] = v + bls[f - N_LAT];
}

// ---------------- layer-2 edge scatter: one wave per edge -------------------
__global__ __launch_bounds__(256)
void k_edge2(const long long* __restrict__ src, const long long* __restrict__ dst,
             int E, const float* __restrict__ H2, const float* __restrict__ dis,
             float* __restrict__ out) {
    int wave = threadIdx.x >> 5;
    int lane = threadIdx.x & 31;
    int e = blockIdx.x * 8 + wave;
    if (e >= E) return;
    int s = (int)src[e], d = (int)dst[e];
    float norm = dis[s] * dis[d];
    const float* hs = H2 + (size_t)s * N_OUT2;
    // it=0 -> mu block, it=1 -> logstd block (lane = latent index)
    atomicAdd(&out[(size_t)d * N_LAT + lane], hs[lane] * norm);
    atomicAdd(&out[(size_t)N_NODES * N_LAT + (size_t)d * N_LAT + lane],
              hs[lane + N_LAT] * norm);
}

// ---------------------------------------------------------------------------
extern "C" void kernel_launch(void* const* d_in, const int* in_sizes, int n_in,
                              void* d_out, int out_size, void* d_ws, size_t ws_size,
                              hipStream_t stream) {
    const float*     x   = (const float*)d_in[0];
    const float*     W1  = (const float*)d_in[1];
    const float*     b1  = (const float*)d_in[2];
    const float*     Wmu = (const float*)d_in[3];
    const float*     bmu = (const float*)d_in[4];
    const float*     Wls = (const float*)d_in[5];
    const float*     bls = (const float*)d_in[6];
    const long long* ei  = (const long long*)d_in[7];

    const int E = in_sizes[7] / 2;
    const long long* src = ei;
    const long long* dst = ei + E;
    float* out = (float*)d_out;

    // workspace carve-up (float units, every region 256B-aligned):
    float* ws  = (float*)d_ws;
    float* deg = ws;                                   //  50048 (rounded)
    float* H1  = deg + 50048;                          //  N*96 fp32
    float* A1  = H1 + (size_t)N_NODES * N_HID;         //  N*96 fp32
    float* H2  = A1 + (size_t)N_NODES * N_HID;         //  N*64 fp32
    float* xbf = H2 + (size_t)N_NODES * N_OUT2;        //  N*512 bf16 (as 12.8M floats)
    float* hbf = xbf + (size_t)N_NODES * N_IN / 2;     //  N*96 bf16
    float* w1t = hbf + (size_t)N_NODES * N_HID / 2;    //  96*512 bf16
    float* wct = w1t + (size_t)N_HID * N_IN / 2;       //  64*96 bf16

    const int T = 256;

    // 0) one-time bf16 conversions
    long long n4 = (long long)N_NODES * N_IN / 4;
    k_cvt_x<<<(unsigned)((n4 + T - 1) / T), T, 0, stream>>>(
        (const float4*)x, (unsigned int*)xbf, n4);
    k_cvt_w1t<<<(N_HID * N_IN + T - 1) / T, T, 0, stream>>>(W1, (__bf16*)w1t);
    k_cvt_wcat<<<N_OUT2, N_HID, 0, stream>>>(Wmu, Wls, (__bf16*)wct);

    // 1) degree + rsqrt
    k_init_deg<<<(N_NODES + T - 1) / T, T, 0, stream>>>(deg);
    k_deg_scatter<<<(E + T - 1) / T, T, 0, stream>>>(dst, E, deg);
    k_rsqrt_inplace<<<(N_NODES + T - 1) / T, T, 0, stream>>>(deg);

    // 2) H1 = x @ W1  (WMMA bf16, LDS-free)
    k_gemm1<<<N_NODES / 16, 192, 0, stream>>>((const __bf16*)xbf, (const __bf16*)w1t, H1);

    // 3) layer-1 aggregation + ReLU->bf16
    k_self1<<<N_NODES, N_HID, 0, stream>>>(H1, deg, b1, A1);
    k_edge1<<<(E + 7) / 8, 256, 0, stream>>>(src, dst, E, H1, deg, A1);
    k_relu_cvt<<<N_NODES, N_HID / 2, 0, stream>>>(A1, (unsigned int*)hbf);

    // 4) H2 = hidden @ [Wmu|Wls]  (WMMA bf16, LDS-free)
    k_gemm2<<<N_NODES / 16, 128, 0, stream>>>((const __bf16*)hbf, (const __bf16*)wct, H2);

    // 5) layer-2 aggregation into d_out (mu | logstd)
    k_self2<<<N_NODES, N_OUT2, 0, stream>>>(H2, deg, bmu, bls, out);
    k_edge2<<<(E + 7) / 8, 256, 0, stream>>>(src, dst, E, H2, deg, out);
}
